// CP_RNN_54365696033390
// MI455X (gfx1250) — matrix operand order repaired and checked by reading, not compiled, not run on hardware
//
#include <hip/hip_runtime.h>
#include <math.h>

#define DI __device__ __forceinline__

typedef __attribute__((ext_vector_type(16))) __bf16        v16bf;
typedef __attribute__((ext_vector_type(8)))  float         v8f;
typedef __attribute__((ext_vector_type(4)))  unsigned int  u32x4;
typedef __attribute__((ext_vector_type(4)))  float         f32x4;

constexpr int N_   = 256;
constexpr int T_   = 2048;
constexpr int IN_  = 6;
constexpr int HID_ = 128;
constexpr int ST_  = 64;
constexpr int LAY_ = 134;   // IN + HID
constexpr int OUT_ = 6;

constexpr int KP   = 168;   // K stride (elements) for K=160 buffers (16B-aligned rows, conflict pad)
constexpr int KPO  = 136;   // K stride for Wo (K=128)
constexpr int NT_G = 144;   // padded N for gated weights (9 tiles of 16)

// ---- LDS layout (ushort units) ----
constexpr int SZ_WTG  = 4 * NT_G * KP;          // Wg1[0],Wg2[0],Wg1[1],Wg2[1] transposed
constexpr int OFF_WTG = 0;
constexpr int OFF_WTA = OFF_WTG + SZ_WTG;       // Wa^T   (128 x KP)
constexpr int OFF_WTB = OFF_WTA + HID_ * KP;    // Wb^T   (128 x KP)
constexpr int OFF_WTO = OFF_WTB + HID_ * KP;    // Wo^T   (16 x KPO)
constexpr int OFF_LVA = OFF_WTO + 16 * KPO;     // Lv buffer A (16 x KP)
constexpr int OFF_LVB = OFF_LVA + 16 * KP;      // Lv buffer B (16 x KP)
constexpr int OFF_NRM = OFF_LVB + 16 * KP;      // 32 floats (double-buffered norms)
constexpr int SMEM_USH = OFF_NRM + 64;
constexpr size_t SMEM_BYTES = (size_t)SMEM_USH * 2;   // ~294.8 KB < 320 KB/WGP

DI __bf16 tobf(float f) { return (__bf16)f; }   // native v_cvt on gfx1250
DI float fast_tanh(float x) {
    float e = __expf(2.0f * x);
    return 1.0f - 2.0f * __frcp_rn(e + 1.0f);   // saturates correctly at +/-1
}
union Frag { u32x4 q[2]; v16bf v; };

// A fragment (16x32 bf16): lane holds row m=l16; runs K=[k0+8h,+8) and [k0+16+8h,+8)
DI v16bf load_a(const unsigned short* lv, int l16, int half, int k0) {
    Frag f;
    const unsigned short* p = lv + l16 * KP + k0 + 8 * half;
    f.q[0] = *(const u32x4*)(p);
    f.q[1] = *(const u32x4*)(p + 16);
    return f.v;
}
// B fragment (32x16 bf16) from transposed weights wt[n][k]: runs K=[k0+8h,+8) and [k0+16+8h,+8)
DI v16bf load_b(const unsigned short* wt, int kp, int n, int half, int k0) {
    Frag f;
    const unsigned short* p = wt + n * kp + k0 + 8 * half;
    f.q[0] = *(const u32x4*)(p);
    f.q[1] = *(const u32x4*)(p + 16);
    return f.v;
}
DI v8f wmma_bf(v16bf a, v16bf b, v8f c) {
    return __builtin_amdgcn_wmma_f32_16x16x32_bf16(false, a, false, b, (short)0, c, false, false);
}
DI v8f splat8(float x) {
    v8f v;
#pragma unroll
    for (int r = 0; r < 8; ++r) v[r] = x;
    return v;
}

__global__ __launch_bounds__(256, 1)
void lmsc_kernel(const float* __restrict__ X,  const float* __restrict__ H0,
                 const float* __restrict__ Wg1,const float* __restrict__ bg1,
                 const float* __restrict__ Wg2,const float* __restrict__ bg2,
                 const float* __restrict__ Wa, const float* __restrict__ ba,
                 const float* __restrict__ Wb, const float* __restrict__ bb,
                 const float* __restrict__ Wh, const float* __restrict__ bh,
                 const float* __restrict__ Wo, const float* __restrict__ bo,
                 float* __restrict__ Y)
{
    extern __shared__ unsigned short smem[];
    unsigned short* wtg = smem + OFF_WTG;
    unsigned short* wta = smem + OFF_WTA;
    unsigned short* wtb = smem + OFF_WTB;
    unsigned short* wto = smem + OFF_WTO;
    unsigned short* lvA = smem + OFF_LVA;
    unsigned short* lvB = smem + OFF_LVB;
    __bf16* lvAb = (__bf16*)lvA;
    __bf16* lvBb = (__bf16*)lvB;
    float* norms = (float*)(smem + OFF_NRM);

    const int tid  = threadIdx.x;
    const int wid  = tid >> 5;
    const int lane = tid & 31;
    const int half = lane >> 4;
    const int l16  = lane & 15;
    const int n0   = wid * 16;            // this wave's output-column tile
    const int b0   = blockIdx.x * 16;     // batch-row tile base

    // ---------------- prologue ----------------
    for (int i = tid; i < 2 * 16 * KP; i += 256) lvA[i] = 0;   // zero lvA + lvB (contiguous)
    __syncthreads();

    // stage H0 tile -> lvB (bf16), Wh^T -> wta (temporarily)
    for (int i = tid; i < 16 * ST_; i += 256) {
        int m = i / ST_, k = i % ST_;
        lvBb[m * KP + k] = tobf(H0[(b0 + m) * ST_ + k]);
    }
    for (int i = tid; i < HID_ * ST_; i += 256) {
        int n = i / ST_, k = i % ST_;
        ((__bf16*)wta)[n * KP + k] = tobf(Wh[k * HID_ + n]);
    }
    __syncthreads();

    // S0 = H0 @ Wh + bh  (each wave owns 16 of 128 hidden columns, kept in registers)
    float Hreg[8];
    {
        v8f acc = splat8(bh[n0 + l16]);
#pragma unroll
        for (int kk = 0; kk < 2; ++kk) {
            v16bf a = load_a(lvB, l16, half, kk * 32);
            v16bf b = load_b(wta, KP, n0 + l16, half, kk * 32);
            acc = wmma_bf(a, b, acc);
        }
#pragma unroll
        for (int r = 0; r < 8; ++r) Hreg[r] = acc[r];
    }
    __syncthreads();   // done reading Wh/H0 staging; safe to overwrite

    // stage gated weights transposed; layer 0 K-rows permuted for H-first Lv layout
    for (int i = tid; i < SZ_WTG; i += 256) {
        int mi  = i / (NT_G * KP);
        int rem = i % (NT_G * KP);
        int n = rem / KP, k = rem % KP;
        int layer = mi >> 1;
        const float* W = (mi & 1) ? Wg2 : Wg1;
        float v = 0.0f;
        if (n < LAY_) {
            int ks = -1;
            if (layer == 0) { if (k < HID_) ks = k + IN_; else if (k < LAY_) ks = k - HID_; }
            else            { if (k < LAY_) ks = k; }
            if (ks >= 0) v = W[(layer * LAY_ + ks) * LAY_ + n];
        }
        ((__bf16*)wtg)[i] = tobf(v);
    }
    for (int i = tid; i < HID_ * KP; i += 256) {
        int n = i / KP, k = i % KP;
        ((__bf16*)wta)[i] = tobf((k < LAY_) ? Wa[k * HID_ + n] : 0.0f);
        ((__bf16*)wtb)[i] = tobf((k < LAY_) ? Wb[k * HID_ + n] : 0.0f);
    }
    for (int i = tid; i < 16 * KPO; i += 256) {
        int n = i / KPO, k = i % KPO;
        ((__bf16*)wto)[i] = tobf((n < OUT_ && k < HID_) ? Wo[k * OUT_ + n] : 0.0f);
    }

    // initial lvA = [H (cols 0..127) | x0/||x0|| (cols 128..133) | zeros]
#pragma unroll
    for (int r = 0; r < 8; ++r)
        lvAb[(r + 8 * half) * KP + n0 + l16] = tobf(Hreg[r]);
    if (tid < 16) {
        const float* xp = X + (size_t)(b0 + tid) * T_ * IN_;
        float s = 0.f, xv[IN_];
#pragma unroll
        for (int i = 0; i < IN_; ++i) { xv[i] = xp[i]; s += xv[i] * xv[i]; }
        float nrm = sqrtf(s);
        norms[tid] = nrm;
        float rn = __frcp_rn(nrm);
#pragma unroll
        for (int i = 0; i < IN_; ++i) lvAb[tid * KP + HID_ + i] = tobf(xv[i] * rn);
    }

    // per-wave bias accumulator-init vectors (loop-invariant)
    const v8f vb_g1l0 = splat8(bg1[0 * LAY_ + n0 + l16]);
    const v8f vb_g2l0 = splat8(bg2[0 * LAY_ + n0 + l16]);
    const v8f vb_g1l1 = splat8(bg1[1 * LAY_ + n0 + l16]);
    const v8f vb_g2l1 = splat8(bg2[1 * LAY_ + n0 + l16]);
    const v8f vb_a    = splat8(ba[n0 + l16]);
    const v8f vb_b    = splat8(bb[n0 + l16]);
    float e1v = 0.f, e2v = 0.f, e3v = 0.f, e4v = 0.f;
    if (wid == 1 && (HID_ + l16) < LAY_) {      // wave 1 also owns tile 8 (cols 128..143)
        e1v = bg1[0 * LAY_ + HID_ + l16];  e2v = bg2[0 * LAY_ + HID_ + l16];
        e3v = bg1[1 * LAY_ + HID_ + l16];  e4v = bg2[1 * LAY_ + HID_ + l16];
    }
    const v8f vb_e1 = splat8(e1v), vb_e2 = splat8(e2v);
    const v8f vb_e3 = splat8(e3v), vb_e4 = splat8(e4v);
    const v8f vb_o  = splat8((wid == 0 && l16 < OUT_) ? bo[l16] : 0.0f);
    __syncthreads();

    // ---------------- recurrence ----------------
#pragma clang loop unroll(disable)
    for (int t = 0; t < T_; ++t) {
        // gated layer 0: lvA -> lvB.  Wave 0 fuses the output projection for the
        // previous step's Hn into the same k-loop, reusing the A fragments.
        v8f g1 = vb_g1l0, g2 = vb_g2l0;
        v8f x1 = vb_e1,   x2 = vb_e2;
        v8f o  = vb_o;
#pragma unroll
        for (int kk = 0; kk < 5; ++kk) {
            const int k0 = kk * 32;
            v16bf a = load_a(lvA, l16, half, k0);
            g1 = wmma_bf(a, load_b(wtg + 0 * NT_G * KP, KP, n0 + l16,  half, k0), g1);
            g2 = wmma_bf(a, load_b(wtg + 1 * NT_G * KP, KP, n0 + l16,  half, k0), g2);
            if (wid == 1) {
                x1 = wmma_bf(a, load_b(wtg + 0 * NT_G * KP, KP, HID_ + l16, half, k0), x1);
                x2 = wmma_bf(a, load_b(wtg + 1 * NT_G * KP, KP, HID_ + l16, half, k0), x2);
            }
            if (wid == 0 && kk < 4)
                o = wmma_bf(a, load_b(wto, KPO, l16, half, k0), o);
        }
        if (wid == 0 && t > 0 && l16 < OUT_) {
#pragma unroll
            for (int r = 0; r < 8; ++r)
                Y[((size_t)(b0 + r + 8 * half) * T_ + (t - 1)) * OUT_ + l16] = o[r];
        }
#pragma unroll
        for (int r = 0; r < 8; ++r)
            lvBb[(r + 8 * half) * KP + n0 + l16] = tobf(fast_tanh(g1[r]) * fast_tanh(g2[r]));
        if (wid == 1) {
#pragma unroll
            for (int r = 0; r < 8; ++r)
                lvBb[(r + 8 * half) * KP + HID_ + l16] = tobf(fast_tanh(x1[r]) * fast_tanh(x2[r]));
        }
        __syncthreads();

        // gated layer 1: lvB -> lvA
        g1 = vb_g1l1; g2 = vb_g2l1;
        x1 = vb_e3;   x2 = vb_e4;
#pragma unroll
        for (int kk = 0; kk < 5; ++kk) {
            const int k0 = kk * 32;
            v16bf a = load_a(lvB, l16, half, k0);
            g1 = wmma_bf(a, load_b(wtg + 2 * NT_G * KP, KP, n0 + l16,  half, k0), g1);
            g2 = wmma_bf(a, load_b(wtg + 3 * NT_G * KP, KP, n0 + l16,  half, k0), g2);
            if (wid == 1) {
                x1 = wmma_bf(a, load_b(wtg + 2 * NT_G * KP, KP, HID_ + l16, half, k0), x1);
                x2 = wmma_bf(a, load_b(wtg + 3 * NT_G * KP, KP, HID_ + l16, half, k0), x2);
            }
        }
#pragma unroll
        for (int r = 0; r < 8; ++r)
            lvAb[(r + 8 * half) * KP + n0 + l16] = tobf(fast_tanh(g1[r]) * fast_tanh(g2[r]));
        if (wid == 1) {
#pragma unroll
            for (int r = 0; r < 8; ++r)
                lvAb[(r + 8 * half) * KP + HID_ + l16] = tobf(fast_tanh(x1[r]) * fast_tanh(x2[r]));
        }
        __syncthreads();

        // alpha/beta projections from lvA; wave 3 prefetches x_{t+1} concurrently
        v8f ga = vb_a, gb = vb_b;
#pragma unroll
        for (int kk = 0; kk < 5; ++kk) {
            const int k0 = kk * 32;
            v16bf a = load_a(lvA, l16, half, k0);
            ga = wmma_bf(a, load_b(wta, KP, n0 + l16, half, k0), ga);
            gb = wmma_bf(a, load_b(wtb, KP, n0 + l16, half, k0), gb);
        }
        float xv0 = 0.f, xv1 = 0.f, xv2 = 0.f, xv3 = 0.f, xv4 = 0.f, xv5 = 0.f, xnrm = 0.f;
        if (wid == 3 && lane < 16 && (t + 1) < T_) {
            const float* xp = X + ((size_t)(b0 + lane) * T_ + (t + 1)) * IN_;
            xv0 = xp[0]; xv1 = xp[1]; xv2 = xp[2]; xv3 = xp[3]; xv4 = xp[4]; xv5 = xp[5];
            xnrm = sqrtf(xv0*xv0 + xv1*xv1 + xv2*xv2 + xv3*xv3 + xv4*xv4 + xv5*xv5);
        }
        __syncthreads();   // all lvA reads complete

        // elementwise state update: Hn = exp(-exp(ga)*norm)*(H - beta) + beta
        const float* nbase = norms + (t & 1) * 16 + 8 * half;
        f32x4 nlo = *(const f32x4*)(nbase);
        f32x4 nhi = *(const f32x4*)(nbase + 4);
#pragma unroll
        for (int r = 0; r < 8; ++r) {
            float alpha = __expf(ga[r]);
            float beta  = fast_tanh(gb[r]);
            float nrm   = (r < 4) ? nlo[r] : nhi[r - 4];
            float Hn = __expf(-alpha * nrm) * (Hreg[r] - beta) + beta;
            Hreg[r] = Hn;
            lvAb[(r + 8 * half) * KP + n0 + l16] = tobf(Hn);
        }
        if (wid == 3 && lane < 16 && (t + 1) < T_) {
            norms[((t + 1) & 1) * 16 + lane] = xnrm;
            float rn = __frcp_rn(xnrm);
            __bf16* row = lvAb + lane * KP + HID_;
            row[0] = tobf(xv0 * rn); row[1] = tobf(xv1 * rn); row[2] = tobf(xv2 * rn);
            row[3] = tobf(xv3 * rn); row[4] = tobf(xv4 * rn); row[5] = tobf(xv5 * rn);
        }
        __syncthreads();
    }

    // epilogue: output projection for final step
    if (wid == 0) {
        v8f o = vb_o;
#pragma unroll
        for (int kk = 0; kk < 4; ++kk) {
            v16bf a = load_a(lvA, l16, half, kk * 32);
            v16bf b = load_b(wto, KPO, l16, half, kk * 32);
            o = wmma_bf(a, b, o);
        }
        if (l16 < OUT_) {
#pragma unroll
            for (int r = 0; r < 8; ++r)
                Y[((size_t)(b0 + r + 8 * half) * T_ + (T_ - 1)) * OUT_ + l16] = o[r];
        }
    }
}

extern "C" void kernel_launch(void* const* d_in, const int* in_sizes, int n_in,
                              void* d_out, int out_size, void* d_ws, size_t ws_size,
                              hipStream_t stream) {
    (void)in_sizes; (void)n_in; (void)out_size; (void)d_ws; (void)ws_size;
    const float* X   = (const float*)d_in[0];
    const float* H0  = (const float*)d_in[1];
    const float* Wg1 = (const float*)d_in[2];
    const float* bg1 = (const float*)d_in[3];
    const float* Wg2 = (const float*)d_in[4];
    const float* bg2 = (const float*)d_in[5];
    const float* Wa  = (const float*)d_in[6];
    const float* ba  = (const float*)d_in[7];
    const float* Wb  = (const float*)d_in[8];
    const float* bb  = (const float*)d_in[9];
    const float* Wh  = (const float*)d_in[10];
    const float* bh  = (const float*)d_in[11];
    const float* Wo  = (const float*)d_in[12];
    const float* bo  = (const float*)d_in[13];
    float* Y = (float*)d_out;

    hipFuncSetAttribute(reinterpret_cast<const void*>(lmsc_kernel),
                        hipFuncAttributeMaxDynamicSharedMemorySize, (int)SMEM_BYTES);
    lmsc_kernel<<<dim3(N_ / 16), dim3(256), SMEM_BYTES, stream>>>(
        X, H0, Wg1, bg1, Wg2, bg2, Wa, ba, Wb, bb, Wh, bh, Wo, bo, Y);
}
